// MeanFieldLayer_57698590654749
// MI455X (gfx1250) — matrix operand order repaired and checked by reading, not compiled, not run on hardware
//
#include <hip/hip_runtime.h>
#include <cstdint>
#include <cstddef>

// ---------------------------------------------------------------------------
// MeanField Bayesian layer, MC-sampled:
//   out[s,b,o] = sum_i x[s,b,i]*(w_mu[i,o] + sigma_w[i,o]*eps_w[s,i,o]) + bias[s,o]
//   sigma_w = exp(0.5*w_log_var),  bias[s,o] = b_mu[o] + exp(0.5*b_lv[o])*eps_b[s,o]
//
// gfx1250 plan:
//  * Pre-pass 1: split x (fp32) into bf16 hi/lo planes in d_ws (done ONCE, not
//    per N-block) -> GEMM A-stage becomes a pure copy, done with
//    global_load_async_to_lds_b128 (ASYNCcnt), double buffered.
//  * Pre-pass 2: sigma_w = exp(0.5*lv) in d_ws -> no transcendentals in hot loop.
//  * Pre-pass 3: fused bias table in d_ws.
//  * Main GEMM: per sample (M=256,N=2048,K=2048), block = full-M 256x64 tile,
//    split-bf16 (hi*hi + hi*lo + lo*hi) v_wmma_f32_16x16x32_bf16 for ~fp32
//    accuracy at bf16-WMMA rate. eps_w (268 MB, the HBM-bound stream) is read
//    nontemporally; w_mu/sigma_w/x-planes stay L2-resident (192 MB L2).
//  * bf16 splits use native __bf16 casts -> v_cvt_pk_bf16_f32 instead of
//    integer-emulated rounding (cuts hot-loop VALU).
// ---------------------------------------------------------------------------

typedef __attribute__((ext_vector_type(16))) __bf16 v16bf;
typedef __attribute__((ext_vector_type(8)))  float  v8f;
typedef __attribute__((ext_vector_type(4)))  float  v4f;   // clang vector: OK for nontemporal builtins
typedef unsigned short ushort_t;

#define S_    16
#define B_    256
#define DIN_  2048
#define DOUT_ 2048

#define BM 256     // full batch: weight tile materialized exactly once
#define BN 64
#define BK 32
#define NK (DIN_ / BK)
#define LDA 40     // padded LDS row stride (ushorts) -> conflict-free b128 reads
#define LDB 40

// d_ws layout
#define SIGW_ELEMS ((size_t)DIN_ * DOUT_)          // fp32 sigma_w
#define X_ELEMS    ((size_t)S_ * B_ * DIN_)        // bf16 hi / lo planes of x
#define BIAS_ELEMS ((size_t)S_ * DOUT_)            // fp32 fused bias

union Frag16 { uint4 q[2]; v16bf v; };
union Pack4  { __bf16 b[4]; uint2 u; };

// ---------------- pre-pass kernels ----------------

__global__ __launch_bounds__(256) void prep_split_x(const float* __restrict__ x,
                                                    ushort_t* __restrict__ Xhi,
                                                    ushort_t* __restrict__ Xlo) {
  const size_t i = ((size_t)blockIdx.x * 256 + threadIdx.x) * 4;
  float4 v = *(const float4*)(x + i);
  float f[4] = {v.x, v.y, v.z, v.w};
  Pack4 hp, lp;
#pragma unroll
  for (int j = 0; j < 4; ++j) {
    __bf16 h = (__bf16)f[j];                 // native RNE cvt
    hp.b[j] = h;
    lp.b[j] = (__bf16)(f[j] - (float)h);     // residual -> lo plane
  }
  *(uint2*)(Xhi + i) = make_uint2(hp.u.x, hp.u.y);
  *(uint2*)(Xlo + i) = make_uint2(lp.u.x, lp.u.y);
}

__global__ __launch_bounds__(256) void prep_sigma(const float* __restrict__ w_lv,
                                                  float* __restrict__ sigW) {
  const size_t i = ((size_t)blockIdx.x * 256 + threadIdx.x) * 4;
  float4 v = *(const float4*)(w_lv + i);
  float4 r;
  r.x = expf(0.5f * v.x); r.y = expf(0.5f * v.y);
  r.z = expf(0.5f * v.z); r.w = expf(0.5f * v.w);
  *(float4*)(sigW + i) = r;
}

__global__ __launch_bounds__(256) void prep_bias(const float* __restrict__ b_mu,
                                                 const float* __restrict__ b_lv,
                                                 const float* __restrict__ eps_b,
                                                 float* __restrict__ biasS) {
  const size_t i = (size_t)blockIdx.x * 256 + threadIdx.x;  // i = s*DOUT + o
  const int o = (int)(i & (DOUT_ - 1));
  biasS[i] = b_mu[o] + expf(0.5f * b_lv[o]) * eps_b[i];
}

// ---------------- main GEMM ----------------

__global__ __launch_bounds__(256) void
meanfield_gemm_wmma(const ushort_t* __restrict__ Xhi,
                    const ushort_t* __restrict__ Xlo,
                    const float*    __restrict__ sigW,
                    const float*    __restrict__ w_mu,
                    const float*    __restrict__ eps_w,
                    const float*    __restrict__ biasS,
                    float*          __restrict__ out)
{
  // Double-buffered LDS: x tile hi/lo [BM][BK], weight tile hi/lo stored
  // N-major [BN][BK] so each lane's B fragment is two 16-byte k-runs.
  __shared__ alignas(16) ushort_t Ah[2][BM][LDA];
  __shared__ alignas(16) ushort_t Al[2][BM][LDA];
  __shared__ alignas(16) ushort_t Bh[2][BN][LDB];
  __shared__ alignas(16) ushort_t Bl[2][BN][LDB];

  const int t     = threadIdx.x;         // 256 threads = 8 waves
  const int lane  = t & 31;
  const int wave  = t >> 5;              // wave owns rows [32*wave, 32*wave+32)
  const int s     = blockIdx.z;
  const int nBase = blockIdx.x * BN;

  const int l15     = lane & 15;
  const int halfsel = (lane >= 16) ? 1 : 0;

  // A-copy role: thread t copies x row t (hi + lo) -> 8 async b128 loads
  const size_t xRowBase = ((size_t)s * B_ + t) * DIN_;
  // B-stage role: thread t materializes 8 weights at k-row (t>>3), cols (t&7)*8
  const int bkk = t >> 3;
  const int bn0 = (t & 7) * 8;

  v8f acc[2][4] = {};

  auto copyA = [&](int buf, int kb) {
    const ushort_t* sH = Xhi + xRowBase + (size_t)kb * BK;
    const ushort_t* sL = Xlo + xRowBase + (size_t)kb * BK;
    unsigned dH = (unsigned)(size_t)&Ah[buf][t][0];
    unsigned dL = (unsigned)(size_t)&Al[buf][t][0];
    // 64B row per plane: 4 async b128 copies; INST_OFFSET applies to both the
    // LDS and global address (ISA 10: async load pseudocode), so one base works.
    asm volatile(
        "global_load_async_to_lds_b128 %0, %2, off\n\t"
        "global_load_async_to_lds_b128 %0, %2, off offset:16\n\t"
        "global_load_async_to_lds_b128 %0, %2, off offset:32\n\t"
        "global_load_async_to_lds_b128 %0, %2, off offset:48\n\t"
        "global_load_async_to_lds_b128 %1, %3, off\n\t"
        "global_load_async_to_lds_b128 %1, %3, off offset:16\n\t"
        "global_load_async_to_lds_b128 %1, %3, off offset:32\n\t"
        "global_load_async_to_lds_b128 %1, %3, off offset:48"
        :: "v"(dH), "v"(dL), "v"(sH), "v"(sL) : "memory");
  };

  auto stageB = [&](int buf, int kb) {
    const int k0 = kb * BK;
    const size_t gi = (size_t)(k0 + bkk) * DOUT_ + nBase + bn0;
    const size_t ei = ((size_t)s * DIN_ + k0 + bkk) * DOUT_ + nBase + bn0;
    float sg[8], mu[8], ep[8];
    *(float4*)&sg[0] = ((const float4*)(sigW + gi))[0];
    *(float4*)&sg[4] = ((const float4*)(sigW + gi))[1];
    *(float4*)&mu[0] = ((const float4*)(w_mu + gi))[0];
    *(float4*)&mu[4] = ((const float4*)(w_mu + gi))[1];
    // eps_w streams 268 MB exactly once -> nontemporal, keep L2 for reused data
    *(v4f*)&ep[0] = __builtin_nontemporal_load((const v4f*)(eps_w + ei));
    *(v4f*)&ep[4] = __builtin_nontemporal_load((const v4f*)(eps_w + ei) + 1);
#pragma unroll
    for (int j = 0; j < 8; ++j) {
      float w = fmaf(sg[j], ep[j], mu[j]);
      __bf16 h = (__bf16)w;                  // native RNE cvt (packs pairwise)
      __bf16 l = (__bf16)(w - (float)h);
      Bh[buf][bn0 + j][bkk] = __builtin_bit_cast(ushort_t, h);
      Bl[buf][bn0 + j][bkk] = __builtin_bit_cast(ushort_t, l);
    }
  };

  // prologue: fill buffer 0
  copyA(0, 0);
  stageB(0, 0);
  asm volatile("s_wait_asynccnt 0" ::: "memory");
  __syncthreads();

  for (int kb = 0; kb < NK; ++kb) {
    const int cur = kb & 1;
    const int nxt = cur ^ 1;

    if (kb + 1 < NK) {       // overlap next tile's traffic with this tile's WMMA
      copyA(nxt, kb + 1);
      stageB(nxt, kb + 1);
    }

    // A fragments (16x32 bf16): lanes 0-15 hold K {0..7,16..23},
    // lanes 16-31 hold K {8..15,24..31} (ISA 7.12.2)
    Frag16 aH[2], aL[2];
#pragma unroll
    for (int m = 0; m < 2; ++m) {
      const int r  = wave * 32 + m * 16 + l15;
      const int ko = halfsel * 8;
      aH[m].q[0] = *(const uint4*)&Ah[cur][r][ko];
      aH[m].q[1] = *(const uint4*)&Ah[cur][r][ko + 16];
      aL[m].q[0] = *(const uint4*)&Al[cur][r][ko];
      aL[m].q[1] = *(const uint4*)&Al[cur][r][ko + 16];
    }

#pragma unroll
    for (int n = 0; n < 4; ++n) {
      // B fragment (32x16 bf16): lanes 0-15 hold K {0..7,16..23} of col N=lane,
      // lanes 16-31 hold K {8..15,24..31} (matches A fragment order)
      const int c  = n * 16 + l15;
      const int ko = halfsel * 8;
      Frag16 bH, bL;
      bH.q[0] = *(const uint4*)&Bh[cur][c][ko];
      bH.q[1] = *(const uint4*)&Bh[cur][c][ko + 16];
      bL.q[0] = *(const uint4*)&Bl[cur][c][ko];
      bL.q[1] = *(const uint4*)&Bl[cur][c][ko + 16];
#pragma unroll
      for (int m = 0; m < 2; ++m) {
        // split-precision: hi*hi + hi*lo + lo*hi (lo*lo ~2^-16 rel, dropped)
        acc[m][n] = __builtin_amdgcn_wmma_f32_16x16x32_bf16(
                        false, aH[m].v, false, bH.v, (short)0, acc[m][n], false, false);
        acc[m][n] = __builtin_amdgcn_wmma_f32_16x16x32_bf16(
                        false, aH[m].v, false, bL.v, (short)0, acc[m][n], false, false);
        acc[m][n] = __builtin_amdgcn_wmma_f32_16x16x32_bf16(
                        false, aL[m].v, false, bH.v, (short)0, acc[m][n], false, false);
      }
    }

    asm volatile("s_wait_asynccnt 0" ::: "memory");
    __syncthreads();
  }

  // epilogue: fused bias + nontemporal stores
  // C/D layout: VGPR v <-> row v + 8*halfsel, col = lane&15 (ISA 7.12.2)
#pragma unroll
  for (int m = 0; m < 2; ++m) {
    const int rowTop = wave * 32 + m * 16 + halfsel * 8;
#pragma unroll
    for (int n = 0; n < 4; ++n) {
      const int col  = nBase + n * 16 + l15;
      const float bi = biasS[(size_t)s * DOUT_ + col];
#pragma unroll
      for (int v = 0; v < 8; ++v) {
        __builtin_nontemporal_store(
            acc[m][n][v] + bi,
            &out[((size_t)s * B_ + rowTop + v) * DOUT_ + col]);
      }
    }
  }
}

// ---------------- host launcher ----------------

extern "C" void kernel_launch(void* const* d_in, const int* in_sizes, int n_in,
                              void* d_out, int out_size, void* d_ws, size_t ws_size,
                              hipStream_t stream) {
  const float* x     = (const float*)d_in[0];
  const float* w_mu  = (const float*)d_in[1];
  const float* w_lv  = (const float*)d_in[2];
  const float* b_mu  = (const float*)d_in[3];
  const float* b_lv  = (const float*)d_in[4];
  const float* eps_w = (const float*)d_in[5];
  const float* eps_b = (const float*)d_in[6];
  float* out = (float*)d_out;

  char* ws = (char*)d_ws;
  float*    sigW = (float*)ws;                       ws += SIGW_ELEMS * sizeof(float);
  ushort_t* Xhi  = (ushort_t*)ws;                    ws += X_ELEMS * sizeof(ushort_t);
  ushort_t* Xlo  = (ushort_t*)ws;                    ws += X_ELEMS * sizeof(ushort_t);
  float*    biasS = (float*)ws;                      // BIAS_ELEMS floats

  prep_split_x<<<dim3((unsigned)(X_ELEMS / 4 / 256)), 256, 0, stream>>>(x, Xhi, Xlo);
  prep_sigma  <<<dim3((unsigned)(SIGW_ELEMS / 4 / 256)), 256, 0, stream>>>(w_lv, sigW);
  prep_bias   <<<dim3((unsigned)(BIAS_ELEMS / 256)), 256, 0, stream>>>(b_mu, b_lv, eps_b, biasS);

  dim3 grid(DOUT_ / BN, 1, S_);   // (32, 1, 16) = 512 workgroups of 8 waves
  meanfield_gemm_wmma<<<grid, 256, 0, stream>>>(Xhi, Xlo, sigW, w_mu, eps_w, biasS, out);
}
